// GAT_37580963840365
// MI455X (gfx1250) — matrix operand order repaired and checked by reading, not compiled, not run on hardware
//
#include <hip/hip_runtime.h>
#include <math.h>

#define HC 128
#define NHEADS 4
#define CCH 32
#define KSTEPS 4      // K = 128 = 4 * 32 for every WMMA GEMM in this net
#define MPW 8         // M-tiles per wave (B-fragment register reuse factor)

typedef __attribute__((ext_vector_type(16))) __bf16 v16bf;
typedef __attribute__((ext_vector_type(8)))  __bf16 v8bf;
typedef __attribute__((ext_vector_type(8)))  float  v8f;

// ---------------- atomic helpers ----------------
__device__ __forceinline__ void atomicMaxF(float* addr, float val) {
  // int/uint ordering trick; location must be initialized to -inf.
  if (val >= 0.0f) atomicMax((int*)addr, __float_as_int(val));
  else             atomicMin((unsigned int*)addr, __float_as_uint(val));
}
__device__ __forceinline__ void atomicAddF(float* addr, float val) {
  // lowers to global_atomic_add_f32 (relaxed, device scope)
  __hip_atomic_fetch_add(addr, val, __ATOMIC_RELAXED, __HIP_MEMORY_SCOPE_AGENT);
}

// ---------------- layer-0 projections (in_dim == 2) ----------------
__global__ void l0_proj_kernel(const float* __restrict__ x,
                               const float* __restrict__ Wq, const float* __restrict__ bq,
                               const float* __restrict__ Wk, const float* __restrict__ bk,
                               const float* __restrict__ Wv, const float* __restrict__ bv,
                               const float* __restrict__ Ws, const float* __restrict__ bs,
                               float* __restrict__ q, float* __restrict__ k,
                               float* __restrict__ v, float* __restrict__ acc, int n_nodes) {
  int idx = blockIdx.x * blockDim.x + threadIdx.x;
  int n = idx >> 7, j = idx & (HC - 1);
  if (n >= n_nodes) return;
  float x0 = x[2 * n], x1 = x[2 * n + 1];
  q[idx]   = fmaf(x0, Wq[j], fmaf(x1, Wq[HC + j], bq[j]));
  k[idx]   = fmaf(x0, Wk[j], fmaf(x1, Wk[HC + j], bk[j]));
  v[idx]   = fmaf(x0, Wv[j], fmaf(x1, Wv[HC + j], bv[j]));
  acc[idx] = fmaf(x0, Ws[j], fmaf(x1, Ws[HC + j], bs[j]));  // skip-connection init
}

// ---------------- softmax state init / fix ----------------
__global__ void init_mden_kernel(float* __restrict__ m, float* __restrict__ den, int n) {
  int i = blockIdx.x * blockDim.x + threadIdx.x;
  if (i < n) { m[i] = -INFINITY; den[i] = 0.0f; }
}
__global__ void mfix_kernel(float* __restrict__ m, int n) {
  int i = blockIdx.x * blockDim.x + threadIdx.x;
  if (i < n) { float mv = m[i]; m[i] = isfinite(mv) ? mv : 0.0f; }
}

// ---------------- edge phase ----------------
// one thread per (edge, head): alpha = <q[dst], k[src]+ea*We> / sqrt(C); atomic-max into m
__global__ void edge_alpha_kernel(const float* __restrict__ q, const float* __restrict__ k,
                                  const int* __restrict__ src, const int* __restrict__ dst,
                                  const float* __restrict__ ea, const float* __restrict__ We,
                                  float* __restrict__ alpha, float* __restrict__ m, int n_edges) {
  int idx = blockIdx.x * blockDim.x + threadIdx.x;
  int e = idx >> 2, h = idx & 3;
  if (e >= n_edges) return;
  int s = src[e], d = dst[e];
  float a = ea[e];
  const float4* qp = (const float4*)(q + d * HC + h * CCH);
  const float4* kp = (const float4*)(k + s * HC + h * CCH);
  const float4* wp = (const float4*)(We + h * CCH);
  float acc = 0.0f;
#pragma unroll
  for (int i = 0; i < 8; ++i) {
    float4 qq = qp[i], kk = kp[i], ww = wp[i];
    acc += qq.x * fmaf(a, ww.x, kk.x) + qq.y * fmaf(a, ww.y, kk.y) +
           qq.z * fmaf(a, ww.z, kk.z) + qq.w * fmaf(a, ww.w, kk.w);
  }
  acc *= 0.17677669529663687f;  // 1/sqrt(32)
  alpha[e * NHEADS + h] = acc;
  atomicMaxF(m + d * NHEADS + h, acc);
}

// ex = exp(alpha - m[dst]); alpha <- ex; den[dst] += ex
__global__ void edge_exp_kernel(float* __restrict__ alpha, const float* __restrict__ m,
                                const int* __restrict__ dst, float* __restrict__ den, int n_edges) {
  int idx = blockIdx.x * blockDim.x + threadIdx.x;
  int e = idx >> 2, h = idx & 3;
  if (e >= n_edges) return;
  int d = dst[e];
  float ex = expf(alpha[e * NHEADS + h] - m[d * NHEADS + h]);
  alpha[e * NHEADS + h] = ex;
  atomicAddF(den + d * NHEADS + h, ex);
}

// w = ex/(den[dst]+eps); acc[dst] += w * (v[src] + ea*We)
__global__ void edge_scatter_kernel(const float* __restrict__ v, const float* __restrict__ alpha,
                                    const float* __restrict__ den,
                                    const int* __restrict__ src, const int* __restrict__ dst,
                                    const float* __restrict__ ea, const float* __restrict__ We,
                                    float* __restrict__ acc, int n_edges) {
  int idx = blockIdx.x * blockDim.x + threadIdx.x;
  int e = idx >> 2, h = idx & 3;
  if (e >= n_edges) return;
  int s = src[e], d = dst[e];
  float a = ea[e];
  float w = alpha[e * NHEADS + h] / (den[d * NHEADS + h] + 1e-16f);
  const float4* vp = (const float4*)(v + s * HC + h * CCH);
  const float4* wp = (const float4*)(We + h * CCH);
  float* op = acc + d * HC + h * CCH;
#pragma unroll
  for (int i = 0; i < 8; ++i) {
    float4 vv = vp[i], ww = wp[i];
    atomicAddF(op + 4 * i + 0, w * fmaf(a, ww.x, vv.x));
    atomicAddF(op + 4 * i + 1, w * fmaf(a, ww.y, vv.y));
    atomicAddF(op + 4 * i + 2, w * fmaf(a, ww.z, vv.z));
    atomicAddF(op + 4 * i + 3, w * fmaf(a, ww.w, vv.w));
  }
}

// ---------------- conversions ----------------
__global__ void relu_to_bf16_kernel(const float* __restrict__ in, __bf16* __restrict__ out, int n) {
  int i = blockIdx.x * blockDim.x + threadIdx.x;
  if (i < n) out[i] = (__bf16)fmaxf(in[i], 0.0f);
}
// in: [K,N] row-major f32 weights -> out: [N,K] row-major bf16 (transposed),
// so B fragments become contiguous 16B loads in the WMMA GEMM.
__global__ void f32_to_bf16_T_kernel(const float* __restrict__ in, __bf16* __restrict__ out,
                                     int K, int N) {
  int i = blockIdx.x * blockDim.x + threadIdx.x;
  if (i >= K * N) return;
  int k = i / N, n = i - k * N;
  out[n * K + k] = (__bf16)in[i];
}

// ---------------- WMMA bf16 GEMM: Out[M,Nn] = A[M,128] @ B[128,Nn] + bias ----------------
// Bt is the transposed bf16 weight [Nn,128]. One wave owns one 16-wide N-tile,
// preloads all 4 K-step B fragments into registers, then streams MPW M-tiles
// through v_wmma_f32_16x16x32_bf16 (fragment layouts per CDNA5 ISA 7.12.2, wave32).
// A fragments are double-buffered (prefetch tile mi+1 before computing tile mi).
// Output tiles are staged through LDS (same-wave DS ops are in-order -> no barrier)
// and written back as coalesced b128 row segments; bf16 mirror is one packed 16B store.
template <bool EMIT_BF16, bool DO_RELU>
__global__ __launch_bounds__(256) void wmma_gemm_bf16_kernel(
    const __bf16* __restrict__ A, const __bf16* __restrict__ Bt,
    const float* __restrict__ bias, float* __restrict__ Out, __bf16* __restrict__ OutBf,
    int M, int Nn) {
  __shared__ float tilebuf[8][256];                  // one 16x16 f32 tile per wave
  const int lane = threadIdx.x;                      // 0..31
  const int ntiles = Nn >> 4;
  const int mtiles = (M + 15) >> 4;
  const int mgroups = (mtiles + MPW - 1) / MPW;
  const int wid = blockIdx.x * blockDim.y + threadIdx.y;  // wave-uniform
  if (wid >= mgroups * ntiles) return;               // uniform exit: EXEC all-ones at WMMA
  const int nt = wid % ntiles;
  const int mg = wid / ntiles;

  const int l15 = lane & 15;
  const bool hi = lane >= 16;
  const int ncol = nt * 16 + l15;                    // output column for this lane
  const float bval = bias[ncol];
  float* tb = tilebuf[threadIdx.y];

  // B fragments: lane holds column ncol, K [kb, kb+8) and [kb+16, kb+24) per step (kb = hi?8:0)
  const __bf16* brow = Bt + (size_t)ncol * HC + (hi ? 8 : 0);
  v16bf bfr[KSTEPS];
#pragma unroll
  for (int kt = 0; kt < KSTEPS; ++kt) {
    v8bf b0 = *(const v8bf*)(brow + kt * 32);
    v8bf b1 = *(const v8bf*)(brow + kt * 32 + 16);
#pragma unroll
    for (int i = 0; i < 8; ++i) { bfr[kt][i] = b0[i]; bfr[kt][8 + i] = b1[i]; }
  }

  // A pointer for tile 0 of this wave's group; advances by 16 rows per tile
  const __bf16* abase = A + ((size_t)(mg * MPW) * 16 + l15) * HC + (hi ? 8 : 0);
  // coalesced store base: lane pair (2k,2k+1) covers row k of the tile
  const int srow = lane >> 1, scol8 = (lane & 1) << 3;
  float* gbase = Out + ((size_t)(mg * MPW) * 16 + srow) * Nn + nt * 16 + scol8;

  v8bf a0[2][KSTEPS], a1[2][KSTEPS];
#pragma unroll
  for (int kt = 0; kt < KSTEPS; ++kt) {              // prefetch tile 0
    a0[0][kt] = *(const v8bf*)(abase + kt * 32);
    a1[0][kt] = *(const v8bf*)(abase + kt * 32 + 16);
  }

#pragma unroll
  for (int mi = 0; mi < MPW; ++mi) {
    const int mt = mg * MPW + mi;
    if (mt >= mtiles) break;                         // wave-uniform
    if (mi + 1 < MPW && mt + 1 < mtiles) {           // prefetch next tile's A frags
      const __bf16* ap = abase + (size_t)(mi + 1) * 16 * HC;
#pragma unroll
      for (int kt = 0; kt < KSTEPS; ++kt) {
        a0[(mi + 1) & 1][kt] = *(const v8bf*)(ap + kt * 32);
        a1[(mi + 1) & 1][kt] = *(const v8bf*)(ap + kt * 32 + 16);
      }
    }
    v8f c = {};
#pragma unroll
    for (int kt = 0; kt < KSTEPS; ++kt) {
      v16bf afrag;
#pragma unroll
      for (int i = 0; i < 8; ++i) {
        afrag[i] = a0[mi & 1][kt][i];
        afrag[8 + i] = a1[mi & 1][kt][i];
      }
      c = __builtin_amdgcn_wmma_f32_16x16x32_bf16(false, afrag, false, bfr[kt],
                                                  (short)0, c, false, false);
    }
    // stage tile in LDS: D layout VGPR r -> tile row r (lanes<16) / r+8 (lanes>=16)
    const int lrow = hi ? 8 : 0;
#pragma unroll
    for (int r = 0; r < 8; ++r) {
      float val = c[r] + bval;
      if (DO_RELU) val = fmaxf(val, 0.0f);
      tb[(lrow + r) * 16 + l15] = val;
    }
    // read back row-major (same-wave LDS in-order) and store coalesced
    const float4* rp = (const float4*)(tb + lane * 8);
    float4 f0 = rp[0], f1 = rp[1];
    float* gp = gbase + (size_t)mi * 16 * Nn;
    ((float4*)gp)[0] = f0;
    ((float4*)gp)[1] = f1;
    if (EMIT_BF16) {
      v8bf ob;
      ob[0] = (__bf16)f0.x; ob[1] = (__bf16)f0.y; ob[2] = (__bf16)f0.z; ob[3] = (__bf16)f0.w;
      ob[4] = (__bf16)f1.x; ob[5] = (__bf16)f1.y; ob[6] = (__bf16)f1.z; ob[7] = (__bf16)f1.w;
      *(v8bf*)(OutBf + ((size_t)(mt * 16 + srow)) * Nn + nt * 16 + scol8) = ob;
    }
  }
}

// ---------------- final head: out[n] = c2[n,:] . Wc3 + bc3 ----------------
__global__ void final_head_kernel(const float* __restrict__ c2, const float* __restrict__ Wc3,
                                  const float* __restrict__ bc3, float* __restrict__ out, int n_nodes) {
  int n = blockIdx.x * blockDim.x + threadIdx.x;
  if (n >= n_nodes) return;
  const float4* cp = (const float4*)(c2 + n * 64);
  const float4* wp = (const float4*)Wc3;
  float s = bc3[0];
#pragma unroll
  for (int i = 0; i < 16; ++i) {
    float4 cc = cp[i], ww = wp[i];
    s += cc.x * ww.x + cc.y * ww.y + cc.z * ww.z + cc.w * ww.w;
  }
  out[n] = s;
}

// ---------------- host orchestration ----------------
extern "C" void kernel_launch(void* const* d_in, const int* in_sizes, int n_in,
                              void* d_out, int out_size, void* d_ws, size_t ws_size,
                              hipStream_t stream) {
  const int N = in_sizes[0] / 2;   // x is [N,2]
  const int E = in_sizes[2];       // edge_attr is [E,1]

  const float* x    = (const float*)d_in[0];
  const int*   ei   = (const int*)d_in[1];
  const int*   src  = ei;
  const int*   dst  = ei + E;
  const float* eatt = (const float*)d_in[2];
  const float *Wq0 = (const float*)d_in[3],  *bq0 = (const float*)d_in[4];
  const float *Wk0 = (const float*)d_in[5],  *bk0 = (const float*)d_in[6];
  const float *Wv0 = (const float*)d_in[7],  *bv0 = (const float*)d_in[8];
  const float *We0 = (const float*)d_in[9];
  const float *Ws0 = (const float*)d_in[10], *bs0 = (const float*)d_in[11];
  const float *Wq1 = (const float*)d_in[12], *bq1 = (const float*)d_in[13];
  const float *Wk1 = (const float*)d_in[14], *bk1 = (const float*)d_in[15];
  const float *Wv1 = (const float*)d_in[16], *bv1 = (const float*)d_in[17];
  const float *We1 = (const float*)d_in[18];
  const float *Ws1 = (const float*)d_in[19], *bs1 = (const float*)d_in[20];
  const float *Wc1 = (const float*)d_in[21], *bc1 = (const float*)d_in[22];
  const float *Wc2 = (const float*)d_in[23], *bc2 = (const float*)d_in[24];
  const float *Wc3 = (const float*)d_in[25], *bc3 = (const float*)d_in[26];

  // ---- workspace layout ----
  char* cur = (char*)d_ws;
  auto take = [&](size_t bytes) -> void* {
    void* p = (void*)cur;
    cur += (bytes + 255) & ~(size_t)255;
    return p;
  };
  const size_t nhc = (size_t)N * HC;
  float* q_buf   = (float*)take(nhc * 4);
  float* k_buf   = (float*)take(nhc * 4);
  float* v_buf   = (float*)take(nhc * 4);
  float* acc_buf = (float*)take(nhc * 4);
  float* alpha   = (float*)take((size_t)E * NHEADS * 4);
  float* m_buf   = (float*)take((size_t)N * NHEADS * 4);
  float* den_buf = (float*)take((size_t)N * NHEADS * 4);
  __bf16* hbf    = (__bf16*)take(nhc * 2);
  __bf16* wq1t = (__bf16*)take(HC * HC * 2);   // all weight mirrors are TRANSPOSED [N,K] bf16
  __bf16* wk1t = (__bf16*)take(HC * HC * 2);
  __bf16* wv1t = (__bf16*)take(HC * HC * 2);
  __bf16* ws1t = (__bf16*)take(HC * HC * 2);
  __bf16* wc1t = (__bf16*)take(HC * HC * 2);
  __bf16* wc2t = (__bf16*)take(HC * 64 * 2);
  // classifier-stage reuse of freed buffers:
  __bf16* h2bf  = (__bf16*)v_buf;   // relu(layer1 out) in bf16
  float*  c1f   = q_buf;            // 128-wide hidden
  __bf16* c1bf  = (__bf16*)k_buf;
  float*  c2f   = acc_buf;          // 64-wide hidden

  const int TB = 256;
  const int gNHC = (int)((nhc + TB - 1) / TB);
  const int gEH  = (E * NHEADS + TB - 1) / TB;
  const int gNH  = (N * NHEADS + TB - 1) / TB;
  const int gN   = (N + TB - 1) / TB;
  dim3 wblk(32, 8);
  auto gemm_grid = [&](int M, int Nn) {
    int mtiles = (M + 15) / 16;
    int mgroups = (mtiles + MPW - 1) / MPW;
    return (mgroups * (Nn / 16) + 7) / 8;
  };

  // ================= layer 0 =================
  l0_proj_kernel<<<gNHC, TB, 0, stream>>>(x, Wq0, bq0, Wk0, bk0, Wv0, bv0, Ws0, bs0,
                                          q_buf, k_buf, v_buf, acc_buf, N);
  init_mden_kernel<<<gNH, TB, 0, stream>>>(m_buf, den_buf, N * NHEADS);
  edge_alpha_kernel<<<gEH, TB, 0, stream>>>(q_buf, k_buf, src, dst, eatt, We0, alpha, m_buf, E);
  mfix_kernel<<<gNH, TB, 0, stream>>>(m_buf, N * NHEADS);
  edge_exp_kernel<<<gEH, TB, 0, stream>>>(alpha, m_buf, dst, den_buf, E);
  edge_scatter_kernel<<<gEH, TB, 0, stream>>>(v_buf, alpha, den_buf, src, dst, eatt, We0, acc_buf, E);
  relu_to_bf16_kernel<<<gNHC, TB, 0, stream>>>(acc_buf, hbf, (int)nhc);

  // weights -> transposed bf16
  const int gW = (HC * HC + TB - 1) / TB;
  f32_to_bf16_T_kernel<<<gW, TB, 0, stream>>>(Wq1, wq1t, HC, HC);
  f32_to_bf16_T_kernel<<<gW, TB, 0, stream>>>(Wk1, wk1t, HC, HC);
  f32_to_bf16_T_kernel<<<gW, TB, 0, stream>>>(Wv1, wv1t, HC, HC);
  f32_to_bf16_T_kernel<<<gW, TB, 0, stream>>>(Ws1, ws1t, HC, HC);
  f32_to_bf16_T_kernel<<<gW, TB, 0, stream>>>(Wc1, wc1t, HC, HC);
  f32_to_bf16_T_kernel<<<(HC * 64 + TB - 1) / TB, TB, 0, stream>>>(Wc2, wc2t, HC, 64);

  // ================= layer 1 (WMMA projections) =================
  wmma_gemm_bf16_kernel<false, false><<<gemm_grid(N, HC), wblk, 0, stream>>>(hbf, wq1t, bq1, q_buf, nullptr, N, HC);
  wmma_gemm_bf16_kernel<false, false><<<gemm_grid(N, HC), wblk, 0, stream>>>(hbf, wk1t, bk1, k_buf, nullptr, N, HC);
  wmma_gemm_bf16_kernel<false, false><<<gemm_grid(N, HC), wblk, 0, stream>>>(hbf, wv1t, bv1, v_buf, nullptr, N, HC);
  wmma_gemm_bf16_kernel<false, false><<<gemm_grid(N, HC), wblk, 0, stream>>>(hbf, ws1t, bs1, acc_buf, nullptr, N, HC);

  init_mden_kernel<<<gNH, TB, 0, stream>>>(m_buf, den_buf, N * NHEADS);
  edge_alpha_kernel<<<gEH, TB, 0, stream>>>(q_buf, k_buf, src, dst, eatt, We1, alpha, m_buf, E);
  mfix_kernel<<<gNH, TB, 0, stream>>>(m_buf, N * NHEADS);
  edge_exp_kernel<<<gEH, TB, 0, stream>>>(alpha, m_buf, dst, den_buf, E);
  edge_scatter_kernel<<<gEH, TB, 0, stream>>>(v_buf, alpha, den_buf, src, dst, eatt, We1, acc_buf, E);
  relu_to_bf16_kernel<<<gNHC, TB, 0, stream>>>(acc_buf, h2bf, (int)nhc);

  // ================= classifier =================
  wmma_gemm_bf16_kernel<true, true><<<gemm_grid(N, HC), wblk, 0, stream>>>(h2bf, wc1t, bc1, c1f, c1bf, N, HC);
  wmma_gemm_bf16_kernel<false, true><<<gemm_grid(N, 64), wblk, 0, stream>>>(c1bf, wc2t, bc2, c2f, nullptr, N, 64);
  final_head_kernel<<<gN, TB, 0, stream>>>(c2f, Wc3, bc3, (float*)d_out, N);
}
